// HardTripletLoss_19902878450408
// MI455X (gfx1250) — matrix operand order repaired and checked by reading, not compiled, not run on hardware
//
#include <hip/hip_runtime.h>
#include <hip/hip_bf16.h>

typedef __attribute__((ext_vector_type(8)))  _Float16 v8h;
typedef __attribute__((ext_vector_type(16))) _Float16 v16h;
typedef __attribute__((ext_vector_type(8)))  float    v8f;

#define N_TOT 4096
#define HALF  2048
#define D     256
#define MARGIN 0.3f
#define EPS    1e-6f
#define INF_BITS 0x7F800000u

union V16U { v16h v; v8h h[2]; };

// ---------------- Kernel 1: L2-normalize rows, emit f16 copy + f32 sq ----------------
__global__ __launch_bounds__(256)
void normalize_kernel(const float* __restrict__ in, _Float16* __restrict__ xh,
                      float* __restrict__ sq) {
    const int lane = threadIdx.x & 31;
    const int wave = threadIdx.x >> 5;
    const int row  = blockIdx.x * 8 + wave;            // 512 blocks * 8 waves = 4096 rows

    const float4* p4 = (const float4*)(in + (size_t)row * D + lane * 8);
    float4 u0 = p4[0], u1 = p4[1];
    float v[8] = {u0.x, u0.y, u0.z, u0.w, u1.x, u1.y, u1.z, u1.w};

    float s = 0.0f;
    #pragma unroll
    for (int q = 0; q < 8; ++q) s += v[q] * v[q];
    #pragma unroll
    for (int m = 16; m >= 1; m >>= 1) s += __shfl_xor(s, m, 32);

    const float inv = 1.0f / (sqrtf(s) + EPS);
    if (lane == 0) sq[row] = s * inv * inv;            // ||xn||^2 in f32, matches reference

    v8h o;
    #pragma unroll
    for (int q = 0; q < 8; ++q) o[q] = (_Float16)(v[q] * inv);
    *(v8h*)(xh + (size_t)row * D + lane * 8) = o;
}

// ---------------- Kernel 2: init atomic-mining arrays ----------------
__global__ __launch_bounds__(256)
void init_kernel(unsigned* __restrict__ rmax, unsigned* __restrict__ rmin) {
    int idx = blockIdx.x * 256 + threadIdx.x;
    if (idx < N_TOT) {
        rmax[idx] = 0u;        // dist_ap: max over positives, 0 if none (reference semantics)
        rmin[idx] = INF_BITS;  // dist_an: min over negatives, patched to 1.0 if none
    }
}

// ---------------- Kernel 3: WMMA cross-Gram + fused hard mining ----------------
// One wave computes a 16(M) x 64(N) strip of G = X1 (2048xD) * X2^T (Dx2048).
// Row reductions -> first-half rows; column reductions -> second-half rows.
__global__ __launch_bounds__(256)
void gemm_reduce_kernel(const _Float16* __restrict__ xh, const float* __restrict__ sq,
                        const int* __restrict__ tg,
                        unsigned* __restrict__ rmax, unsigned* __restrict__ rmin) {
    const int lane = threadIdx.x & 31;
    const int wave = threadIdx.x >> 5;
    const int wg   = blockIdx.x * 8 + wave;            // 512 blocks * 8 waves = 4096 tiles
    const int tileM = wg & 127;                        // 128 tiles of 16 rows
    const int tileN = wg >> 7;                         // 32 tiles of 64 cols
    const int lm = lane & 15;
    const int hi = lane >> 4;

    const _Float16* x1 = xh;                           // rows 0..2047
    const _Float16* x2 = xh + (size_t)HALF * D;        // rows 2048..4095

    const int rA = tileM * 16;
    const int cB = tileN * 64;

    const _Float16* arow  = x1 + (size_t)(rA + lm) * D;
    const _Float16* brow0 = x2 + (size_t)(cB + lm) * D;   // tile t adds t*16*D

    v8f acc[4] = {v8f{}, v8f{}, v8f{}, v8f{}};

    #pragma unroll
    for (int k0 = 0; k0 < D; k0 += 32) {
        // A f16, contiguous k order (as B): lanes 0-15 hold K=k0..k0+15, lanes 16-31 K=k0+16..k0+31.
        V16U a;
        a.v = *(const v16h*)(arow + k0 + hi * 16);
        #pragma unroll
        for (int t = 0; t < 4; ++t) {
            // B f16, contiguous k order (as A): lanes 0-15 hold K=k0..k0+15, lanes 16-31 K=k0+16..k0+31.
            V16U b;
            b.v = *(const v16h*)(brow0 + (size_t)t * 16 * D + k0 + hi * 16);
            acc[t] = __builtin_amdgcn_wmma_f32_16x16x32_f16(
                false, a.v, false, b.v, (short)0, acc[t], false, false);
        }
    }

    const float inf = __uint_as_float(INF_BITS);

    // Column metadata (fixed per lane, per tile t)
    int jc[4]; float sqB[4]; int tB[4];
    #pragma unroll
    for (int t = 0; t < 4; ++t) {
        jc[t]  = cB + t * 16 + lm;
        sqB[t] = sq[HALF + jc[t]];
        tB[t]  = tg[HALF + jc[t]];
    }

    float colAp[4] = {0.f, 0.f, 0.f, 0.f};
    float colAn[4] = {inf, inf, inf, inf};

    #pragma unroll
    for (int v = 0; v < 8; ++v) {
        const int   i   = rA + v + hi * 8;             // C layout: VGPR v -> M=v (+8 for hi lanes)
        const float sqA = sq[i];
        const int   tA  = tg[i];

        float rAp = 0.0f, rAn = inf;
        #pragma unroll
        for (int t = 0; t < 4; ++t) {
            const float g  = acc[t][v];
            const float d  = sqrtf(fmaxf(sqA + sqB[t] - 2.0f * g, EPS));
            const bool  mk = (tA == tB[t]);
            const float ap = mk ? d : 0.0f;
            const float an = mk ? inf : d;
            rAp = fmaxf(rAp, ap);
            rAn = fminf(rAn, an);
            colAp[t] = fmaxf(colAp[t], ap);
            colAn[t] = fminf(colAn[t], an);
        }
        // row reduce over the 64 columns held by this half-wave
        #pragma unroll
        for (int m = 8; m >= 1; m >>= 1) {
            rAp = fmaxf(rAp, __shfl_xor(rAp, m, 32));
            rAn = fminf(rAn, __shfl_xor(rAn, m, 32));
        }
        if (lm == 0) {   // lanes 0 and 16 cover rows v and v+8
            atomicMax(&rmax[i], __float_as_uint(rAp));
            atomicMin(&rmin[i], __float_as_uint(rAn));
        }
    }

    // column reduce: combine the two 8-row halves (lanes L <-> L^16)
    #pragma unroll
    for (int t = 0; t < 4; ++t) {
        float cAp = fmaxf(colAp[t], __shfl_xor(colAp[t], 16, 32));
        float cAn = fminf(colAn[t], __shfl_xor(colAn[t], 16, 32));
        if (hi == 0) {
            atomicMax(&rmax[HALF + jc[t]], __float_as_uint(cAp));
            atomicMin(&rmin[HALF + jc[t]], __float_as_uint(cAn));
        }
    }
}

// ---------------- Kernel 4: deterministic final reduction ----------------
__global__ __launch_bounds__(256)
void finish_kernel(const unsigned* __restrict__ rmax, const unsigned* __restrict__ rmin,
                   float* __restrict__ out) {
    __shared__ float sm[256];
    const int tid = threadIdx.x;
    float acc = 0.0f;
    for (int r = tid; r < N_TOT; r += 256) {           // fixed order -> deterministic
        const float    ap = __uint_as_float(rmax[r]);
        const unsigned mu = rmin[r];
        const float    an = (mu >= INF_BITS) ? 1.0f : __uint_as_float(mu);
        acc += fmaxf(ap - an + MARGIN, 0.0f);
    }
    sm[tid] = acc;
    __syncthreads();
    for (int s = 128; s > 0; s >>= 1) {
        if (tid < s) sm[tid] += sm[tid + s];
        __syncthreads();
    }
    if (tid == 0) out[0] = sm[0] / (float)N_TOT;
}

extern "C" void kernel_launch(void* const* d_in, const int* in_sizes, int n_in,
                              void* d_out, int out_size, void* d_ws, size_t ws_size,
                              hipStream_t stream) {
    const float* in = (const float*)d_in[0];   // (4096, 256) f32
    const int*   tg = (const int*)d_in[1];     // (4096,) int
    float*       out = (float*)d_out;          // scalar f32

    _Float16* xh   = (_Float16*)d_ws;                          // 4096*256 f16 (2 MB)
    float*    sq   = (float*)(xh + (size_t)N_TOT * D);         // 4096 f32
    unsigned* rmax = (unsigned*)(sq + N_TOT);                  // 4096 u32
    unsigned* rmin = rmax + N_TOT;                             // 4096 u32

    normalize_kernel<<<512, 256, 0, stream>>>(in, xh, sq);
    init_kernel<<<16, 256, 0, stream>>>(rmax, rmin);
    gemm_reduce_kernel<<<512, 256, 0, stream>>>(xh, sq, tg, rmax, rmin);
    finish_kernel<<<1, 256, 0, stream>>>(rmax, rmin, out);
}
